// geom_graph_attn_76124000354509
// MI455X (gfx1250) — matrix operand order repaired and checked by reading, not compiled, not run on hardware
//
#include <hip/hip_runtime.h>

// ---------------- problem constants ----------------
constexpr int kInDim = 20;
constexpr int kDim   = 128;
constexpr int kH     = 8;
constexpr int kL     = 3;
constexpr int kB     = 2;
constexpr int kN     = 192;      // sequence length == pair H/W
constexpr int kDH    = 16;
constexpr int kC2    = 256;      // 2*kDim pair channels
constexpr int kNPIX  = kN * kN;  // 36864
constexpr int kPW    = kN + 2;   // padded width 194
constexpr int kPPIX  = kPW * kPW;// 37636
constexpr int kSTEPS = 9 * (kC2 / 32); // 72 WMMA K-steps for conv2d
constexpr float kEPS = 1e-5f;

// WMMA vector types (CDNA5, wave32)
typedef __attribute__((ext_vector_type(16))) __bf16        v16bf;
typedef __attribute__((ext_vector_type(8)))  float         v8f;
typedef __attribute__((ext_vector_type(8)))  unsigned int  v8u;

__device__ __forceinline__ unsigned short f2bf(float f) {
  unsigned int u = __float_as_uint(f);
  u = (u + 0x7FFFu + ((u >> 16) & 1u)) >> 16;   // round-to-nearest-even
  return (unsigned short)u;
}

// two aligned 16B loads -> one 16x bf16 WMMA fragment
__device__ __forceinline__ v16bf ldpair(const void* p0, const void* p1) {
  uint4 x = *(const uint4*)p0;
  uint4 y = *(const uint4*)p1;
  v8u t;
  t[0] = x.x; t[1] = x.y; t[2] = x.z; t[3] = x.w;
  t[4] = y.x; t[5] = y.y; t[6] = y.z; t[7] = y.w;
  return __builtin_bit_cast(v16bf, t);
}

// ---------------- input conv1d (both stems) ----------------
__global__ void k_conv_in(const float* __restrict__ seq,
                          const float* __restrict__ w1,
                          const float* __restrict__ w2,
                          float* __restrict__ x, float* __restrict__ c) {
  int t = blockIdx.x * 256 + threadIdx.x;
  const int total = kB * kDim * kN;
  if (t >= 2 * total) return;
  int which = t / total;
  int r = t % total;
  int n = r % kN;
  int co = (r / kN) % kDim;
  int b = r / (kN * kDim);
  const float* w = which ? w2 : w1;
  float s = 0.f;
  for (int ci = 0; ci < kInDim; ++ci)
    for (int tt = 0; tt < 3; ++tt) {
      int nn = n + tt - 1;
      if (nn >= 0 && nn < kN)
        s += seq[(b * kN + nn) * kInDim + ci] * w[(co * kInDim + ci) * 3 + tt];
    }
  if (which) c[(b * kDim + co) * kN + n] = s;
  else       x[(b * kN + n) * kDim + co] = s;
}

// ---------------- per-layer conv1d (DIM->DIM, k=3, pad 1) ----------------
__global__ void k_conv1d(const float* __restrict__ cin,
                         const float* __restrict__ w,
                         float* __restrict__ out) {
  int t = blockIdx.x * 256 + threadIdx.x;
  if (t >= kB * kDim * kN) return;
  int n = t % kN;
  int co = (t / kN) % kDim;
  int b = t / (kN * kDim);
  float s = 0.f;
  for (int ci = 0; ci < kDim; ++ci) {
    const float* row = cin + (b * kDim + ci) * kN;
    const float* wr = w + (co * kDim + ci) * 3;
    if (n > 0)      s += row[n - 1] * wr[0];
    s += row[n] * wr[1];
    if (n < kN - 1) s += row[n + 1] * wr[2];
  }
  out[(b * kDim + co) * kN + n] = s;
}

// ---------------- BN over (B, N) per channel, fused apply+relu ----------------
__global__ void k_bn1(const float* __restrict__ ctmp,
                      const float* __restrict__ g, const float* __restrict__ beta,
                      float* __restrict__ cout) {
  int co = blockIdx.x;
  int tid = threadIdx.x; // 128
  __shared__ float ss[128], sq[128];
  float s = 0.f, q = 0.f;
  for (int e = tid; e < kB * kN; e += 128) {
    int b = e / kN, n = e % kN;
    float v = ctmp[(b * kDim + co) * kN + n];
    s += v; q += v * v;
  }
  ss[tid] = s; sq[tid] = q;
  __syncthreads();
  for (int st = 64; st > 0; st >>= 1) {
    if (tid < st) { ss[tid] += ss[tid + st]; sq[tid] += sq[tid + st]; }
    __syncthreads();
  }
  float m = ss[0] / (kB * kN);
  float var = sq[0] / (kB * kN) - m * m;
  float istd = rsqrtf(var + kEPS);
  float gg = g[co], bb = beta[co];
  for (int e = tid; e < kB * kN; e += 128) {
    int b = e / kN, n = e % kN;
    float v = ctmp[(b * kDim + co) * kN + n];
    cout[(b * kDim + co) * kN + n] = fmaxf((v - m) * istd * gg + bb, 0.f);
  }
}

// ---------------- x = LN(x + c^T) ----------------
__global__ void k_ln(float* __restrict__ x, const float* __restrict__ c,
                     const float* __restrict__ g, const float* __restrict__ beta) {
  int row = blockIdx.x;           // b*N + n
  int d = threadIdx.x;            // 128
  int b = row / kN, n = row % kN;
  float t = x[(size_t)row * kDim + d] + c[((size_t)b * kDim + d) * kN + n];
  __shared__ float ss[128], sq[128];
  ss[d] = t; sq[d] = t * t;
  __syncthreads();
  for (int st = 64; st > 0; st >>= 1) {
    if (d < st) { ss[d] += ss[d + st]; sq[d] += sq[d + st]; }
    __syncthreads();
  }
  float m = ss[0] / kDim;
  float var = sq[0] / kDim - m * m;
  float istd = rsqrtf(var + kEPS);
  x[(size_t)row * kDim + d] = (t - m) * istd * g[d] + beta[d];
}

// ---------------- zero padded bf16 buffer (halo stays 0 forever) ----------------
__global__ void k_zero32(unsigned int* __restrict__ p, size_t nWords) {
  size_t t = (size_t)blockIdx.x * 256 + threadIdx.x;
  if (t < nWords) p[t] = 0u;
}

// ---- c2 (NHWC f32) init / accumulate seq2pair; emit padded-NHWC bf16 copy ----
__global__ void k_pair(float* __restrict__ c2, unsigned short* __restrict__ c2bf,
                       const float* __restrict__ c, int mode) {
  size_t t = (size_t)blockIdx.x * 256 + threadIdx.x;   // ((b*N+i)*N+j)*C2+ch
  if (t >= (size_t)kB * kNPIX * kC2) return;
  int ch = (int)(t % kC2);
  size_t r = t / kC2;
  int j = (int)(r % kN); r /= kN;
  int i = (int)(r % kN);
  int b = (int)(r / kN);
  float add = (ch < kDim) ? c[((size_t)b * kDim + ch) * kN + i]
                          : c[((size_t)b * kDim + (ch - kDim)) * kN + j];
  if (mode) {
    float v = c2[t] + add;
    c2[t] = v;
    c2bf[(((size_t)b * kPW + (i + 1)) * kPW + (j + 1)) * kC2 + ch] = f2bf(v);
  } else {
    c2[t] = add;
  }
}

// ---------------- conv2d as implicit GEMM with WMMA bf16 ----------------
// input: padded NHWC bf16 (B,194,194,256); weights packed step-major:
//   wpk[step*C2*16 + co*16 + kh*8 + vv] = {bf16(k0), bf16(k0+1)},
//   k0 = step*32 + kh*16 + vv*2, k = tap*256 + cin (tap-major).
// Each wave computes TWO adjacent 16x16 output tiles (co, co+16): A loaded once.
// out: NHWC f32 (for BN stats)
__global__ void __launch_bounds__(256)
k_conv2_wmma(const unsigned short* __restrict__ c2bf,
             const unsigned int* __restrict__ wpk,
             float* __restrict__ outNHWC) {
  int wave = blockIdx.x * 8 + (threadIdx.x >> 5);
  int lane = threadIdx.x & 31;
  int npair = wave & 7;                       // 8 pairs of N-tiles (32 cols each)
  int mtile = (wave >> 3) % (kNPIX / 16);
  int b = (wave >> 3) / (kNPIX / 16);
  int ml = lane & 15;
  int kh = lane >> 4;
  int m = mtile * 16 + ml;
  int oy = m / kN, ox = m % kN;
  int co0 = npair * 32 + ml;                  // tile 0 column; tile 1 = co0+16
  const unsigned short* inb = c2bf + (size_t)b * kC2 * kPPIX + kh * 16;
  const unsigned int* bp = wpk + co0 * 16 + kh * 8;   // tile1 at +256 u32
  v8f acc0 = {};
  v8f acc1 = {};
#pragma unroll
  for (int tap = 0; tap < 9; ++tap) {
    const int ty = tap / 3, tx = tap % 3;     // constants after unroll
    const unsigned short* pp = inb + ((size_t)(oy + ty) * kPW + (ox + tx)) * kC2;
#pragma unroll
    for (int s = 0; s < kC2 / 32; ++s) {
      v16bf a = ldpair(pp + s * 32, pp + s * 32 + 8);
      const unsigned int* bq = bp + (size_t)(tap * (kC2 / 32) + s) * (kC2 * 16);
      v16bf bm0 = ldpair(bq, bq + 4);
      v16bf bm1 = ldpair(bq + 256, bq + 260);
      acc0 = __builtin_amdgcn_wmma_f32_16x16x32_bf16(false, a, false, bm0,
                                                     (short)0, acc0, false, false);
      acc1 = __builtin_amdgcn_wmma_f32_16x16x32_bf16(false, a, false, bm1,
                                                     (short)0, acc1, false, false);
    }
  }
#pragma unroll
  for (int r = 0; r < 8; ++r) {
    int mm = mtile * 16 + r + 8 * kh;
    size_t base = ((size_t)b * kNPIX + mm) * kC2;
    outNHWC[base + co0]      = acc0[r];
    outNHWC[base + co0 + 16] = acc1[r];
  }
}

// ---------------- BN over (B,H,W) per channel, NHWC -> NHWC, relu ----------------
__global__ void k_bn2(const float* __restrict__ nhwc,
                      const float* __restrict__ g, const float* __restrict__ beta,
                      float* __restrict__ c2) {
  int co = blockIdx.x;       // 256 channels
  int tid = threadIdx.x;     // 256
  const int TOTP = kB * kNPIX;
  __shared__ float ss[256], sq[256];
  float s = 0.f, q = 0.f;
  for (int p = tid; p < TOTP; p += 256) {
    float v = nhwc[(size_t)p * kC2 + co];
    s += v; q += v * v;
  }
  ss[tid] = s; sq[tid] = q;
  __syncthreads();
  for (int st = 128; st > 0; st >>= 1) {
    if (tid < st) { ss[tid] += ss[tid + st]; sq[tid] += sq[tid + st]; }
    __syncthreads();
  }
  float m = ss[0] / TOTP;
  float var = sq[0] / TOTP - m * m;
  float istd = rsqrtf(var + kEPS);
  float gg = g[co], bb = beta[co];
  for (int p = tid; p < TOTP; p += 256) {
    float v = nhwc[(size_t)p * kC2 + co];
    c2[(size_t)p * kC2 + co] = fmaxf((v - m) * istd * gg + bb, 0.f);
  }
}

// ---------------- gate = sigmoid(einsum(dsym, wd) + b), c2 in NHWC ----------------
__global__ void k_gate(const float* __restrict__ c2,
                       const float* __restrict__ wd, const float* __restrict__ wdb,
                       float* __restrict__ gate) {
  size_t t = (size_t)blockIdx.x * 256 + threadIdx.x;
  if (t >= (size_t)kB * kH * kN * kN) return;
  int j = (int)(t % kN);
  size_t r = t / kN;
  int i = (int)(r % kN); r /= kN;
  int h = (int)(r % kH);
  int b = (int)(r / kH);
  const float* pij = c2 + ((size_t)b * kNPIX + i * kN + j) * kC2;
  const float* pji = c2 + ((size_t)b * kNPIX + j * kN + i) * kC2;
  const float* wr = wd + h * kC2;
  float s = 0.f;
  for (int cc = 0; cc < kC2; ++cc) s += (pij[cc] + pji[cc]) * wr[cc];
  s += wdb[h];
  gate[t] = 1.f / (1.f + __expf(-s));
}

// ---------------- f32 (+opt add) -> bf16 ----------------
__global__ void k_tobf(const float* __restrict__ a, const float* __restrict__ b,
                       unsigned short* __restrict__ dst, int n, int hasb) {
  int t = blockIdx.x * 256 + threadIdx.x;
  if (t >= n) return;
  float v = a[t];
  if (hasb) v += b[t];
  dst[t] = f2bf(v);
}

// ------- generic WMMA GEMM: out = relu?(A[384x128] @ B[128x128] + bias) -------
// wpk[s*128*16 + co*16 + kh*8 + vv] = {bf16(k0), bf16(k0+1)}, k0 = s*32+kh*16+vv*2
__global__ void __launch_bounds__(256)
k_gemm_wmma(const unsigned short* __restrict__ Abf,
            const unsigned int* __restrict__ wpk,
            const float* __restrict__ bias,
            float* __restrict__ out, int relu) {
  int wave = blockIdx.x * 8 + (threadIdx.x >> 5);
  int lane = threadIdx.x & 31;
  int ntile = wave & 7;        // 8 N-tiles (128 cols)
  int mtile = wave >> 3;       // 24 M-tiles (384 rows)
  int ml = lane & 15;
  int kh = lane >> 4;
  int co = ntile * 16 + ml;
  const unsigned short* arow = Abf + (size_t)(mtile * 16 + ml) * kDim + kh * 16;
  const unsigned int* bp = wpk + co * 16 + kh * 8;
  v8f acc = {};
#pragma unroll
  for (int s = 0; s < kDim / 32; ++s) {
    v16bf a = ldpair(arow + s * 32, arow + s * 32 + 8);
    const unsigned int* bq = bp + (size_t)s * (kDim * 16);
    v16bf bm = ldpair(bq, bq + 4);
    acc = __builtin_amdgcn_wmma_f32_16x16x32_bf16(false, a, false, bm,
                                                  (short)0, acc, false, false);
  }
#pragma unroll
  for (int r = 0; r < 8; ++r) {
    int mm = mtile * 16 + r + 8 * kh;
    float v = acc[r] + bias[co];
    if (relu) v = fmaxf(v, 0.f);
    out[(size_t)mm * kDim + co] = v;
  }
}

// ---------------- attention: scores, masked softmax, gate, weighted V ----------------
__global__ void k_attn(const float* __restrict__ q, const float* __restrict__ k,
                       const float* __restrict__ v, const float* __restrict__ gate,
                       const float* __restrict__ mask, float* __restrict__ upd) {
  int blk = blockIdx.x;              // ((b*H)+h)*N + i
  int i = blk % kN;
  int h = (blk / kN) % kH;
  int b = blk / (kN * kH);
  int j = threadIdx.x;               // 192
  __shared__ float sa[kN], sr[kN];
  const float* qrow = q + ((size_t)b * kN + i) * kDim + h * kDH;
  const float* krow = k + ((size_t)b * kN + j) * kDim + h * kDH;
  float a = 0.f;
#pragma unroll
  for (int d = 0; d < kDH; ++d) a += qrow[d] * krow[d];
  a *= 0.25f;                        // 1/sqrt(16)
  sr[j] = a;
  __syncthreads();
  for (int s = 128; s > 0; s >>= 1) {
    if (j < s && j + s < kN) sr[j] = fmaxf(sr[j], sr[j + s]);
    __syncthreads();
  }
  float mx = sr[0];
  __syncthreads();
  float e = __expf(a - mx) * mask[b * kN + j];
  sr[j] = e;
  __syncthreads();
  for (int s = 128; s > 0; s >>= 1) {
    if (j < s && j + s < kN) sr[j] += sr[j + s];
    __syncthreads();
  }
  float denom = sr[0] + 1e-6f;
  __syncthreads();
  sa[j] = (e / denom) * gate[(((size_t)b * kH + h) * kN + i) * kN + j];
  __syncthreads();
  if (j < kDH) {
    float acc = 0.f;
    for (int jj = 0; jj < kN; ++jj)
      acc += sa[jj] * v[((size_t)b * kN + jj) * kDim + h * kDH + j];
    upd[((size_t)b * kN + i) * kDim + h * kDH + j] = acc;
  }
}

// ---------------- weight packers (step-major B layout, K-pairs in u32) ----------------
__global__ void k_pack_conv(const float* __restrict__ w, unsigned int* __restrict__ dst) {
  size_t t = (size_t)blockIdx.x * 256 + threadIdx.x;   // over L * 72 * 256 * 16
  const size_t per = (size_t)kSTEPS * kC2 * 16;        // 294912
  if (t >= (size_t)kL * per) return;
  int l   = (int)(t / per);
  size_t r = t % per;
  int vv  = (int)(r & 7);
  int kh  = (int)((r >> 3) & 1);
  int co  = (int)((r >> 4) & (kC2 - 1));
  int stp = (int)(r >> 12);                            // 0..71
  int k0  = stp * 32 + kh * 16 + vv * 2;
  int tap = k0 >> 8;
  int cin = k0 & 255;
  int ky = tap / 3, kx = tap % 3;
  const float* wl = w + (size_t)l * kC2 * kC2 * 9;
  float v0 = wl[(((size_t)co * kC2 + cin) * 3 + ky) * 3 + kx];
  float v1 = wl[(((size_t)co * kC2 + cin + 1) * 3 + ky) * 3 + kx];
  dst[(size_t)l * per + r] = (unsigned int)f2bf(v0) | ((unsigned int)f2bf(v1) << 16);
}

__global__ void k_pack_lin(const float* __restrict__ w, unsigned int* __restrict__ dst) {
  int t = blockIdx.x * 256 + threadIdx.x;  // 4 * 128 * 16 = 8192
  if (t >= (kDim / 32) * kDim * 16) return;
  int vv = t & 7;
  int kh = (t >> 3) & 1;
  int co = (t >> 4) & (kDim - 1);
  int s  = t >> 11;                        // 0..3
  int k0 = s * 32 + kh * 16 + vv * 2;      // B = W^T
  unsigned int out = (unsigned int)f2bf(w[(size_t)co * kDim + k0]) |
                     ((unsigned int)f2bf(w[(size_t)co * kDim + k0 + 1]) << 16);
  dst[t] = out;
}

// ==================== host launcher ====================
extern "C" void kernel_launch(void* const* d_in, const int* in_sizes, int n_in,
                              void* d_out, int out_size, void* d_ws, size_t ws_size,
                              hipStream_t stream) {
  (void)in_sizes; (void)n_in; (void)out_size; (void)ws_size;
  const float* seq   = (const float*)d_in[0];
  const float* maskp = (const float*)d_in[1];
  const float* c1w   = (const float*)d_in[2];
  const float* c2w   = (const float*)d_in[3];
  const float* lng   = (const float*)d_in[4];
  const float* lnb   = (const float*)d_in[5];
  const float* c1dw  = (const float*)d_in[6];
  const float* bn1g  = (const float*)d_in[7];
  const float* bn1b  = (const float*)d_in[8];
  const float* c2dw  = (const float*)d_in[9];
  const float* bn2g  = (const float*)d_in[10];
  const float* bn2b  = (const float*)d_in[11];
  const float* wqw   = (const float*)d_in[12];
  const float* wqb   = (const float*)d_in[13];
  const float* wkw   = (const float*)d_in[14];
  const float* wkb   = (const float*)d_in[15];
  const float* wvw   = (const float*)d_in[16];
  const float* wvb   = (const float*)d_in[17];
  const float* wdw   = (const float*)d_in[18];
  const float* wdb   = (const float*)d_in[19];
  const float* wlw   = (const float*)d_in[20];
  const float* wlb   = (const float*)d_in[21];
  const float* fcw   = (const float*)d_in[22];
  const float* fcb   = (const float*)d_in[23];

  // ---- workspace partition ----
  char* ws = (char*)d_ws;
  size_t off = 0;
  auto take = [&](size_t bytes) -> void* {
    void* p = ws + off;
    off = (off + bytes + 255) & ~(size_t)255;
    return p;
  };
  const size_t NX = (size_t)kB * kN * kDim;         // 49152
  float* x_   = (float*)take(NX * 4);
  float* c_   = (float*)take(NX * 4);
  float* ct_  = (float*)take(NX * 4);
  float* xq   = (float*)take(NX * 4);
  float* xk   = (float*)take(NX * 4);
  float* xv   = (float*)take(NX * 4);
  float* updb = (float*)take(NX * 4);
  float* gate = (float*)take((size_t)kB * kH * kN * kN * 4);
  unsigned short* xbf = (unsigned short*)take(NX * 2);
  float* c2   = (float*)take((size_t)kB * kNPIX * kC2 * 4);    // NHWC
  float* c2t  = (float*)take((size_t)kB * kNPIX * kC2 * 4);    // NHWC conv out
  unsigned short* c2bf = (unsigned short*)take((size_t)kB * kPPIX * kC2 * 2); // padded NHWC
  const size_t wpkcPer = (size_t)kSTEPS * kC2 * 16;            // 294912 u32
  unsigned int* wpkc = (unsigned int*)take((size_t)kL * wpkcPer * 4);
  unsigned int* wpkl = (unsigned int*)take((size_t)13 * 8192 * 4);

  // ---- one-time prep (deterministic, re-run every call) ----
  k_conv_in<<<(2 * (int)NX + 255) / 256, 256, 0, stream>>>(seq, c1w, c2w, x_, c_);
  k_pack_conv<<<(int)((kL * wpkcPer + 255) / 256), 256, 0, stream>>>(c2dw, wpkc);
  for (int l = 0; l < kL; ++l) {
    k_pack_lin<<<32, 256, 0, stream>>>(wqw + (size_t)l * kDim * kDim, wpkl + (l * 4 + 0) * 8192);
    k_pack_lin<<<32, 256, 0, stream>>>(wkw + (size_t)l * kDim * kDim, wpkl + (l * 4 + 1) * 8192);
    k_pack_lin<<<32, 256, 0, stream>>>(wvw + (size_t)l * kDim * kDim, wpkl + (l * 4 + 2) * 8192);
    k_pack_lin<<<32, 256, 0, stream>>>(wlw + (size_t)l * kDim * kDim, wpkl + (l * 4 + 3) * 8192);
  }
  k_pack_lin<<<32, 256, 0, stream>>>(fcw, wpkl + 12 * 8192);

  // zero the padded bf16 buffer once; only interior is ever rewritten
  {
    size_t words = (size_t)kB * kPPIX * kC2 / 2;   // ushort pairs
    k_zero32<<<(int)((words + 255) / 256), 256, 0, stream>>>((unsigned int*)c2bf, words);
  }

  const int pairBlocks = (int)(((size_t)kB * kNPIX * kC2 + 255) / 256);
  k_pair<<<pairBlocks, 256, 0, stream>>>(c2, c2bf, c_, 0);   // c2 = seq2pair(c0)

  const int convBlocks = kB * (kNPIX / 16) * (kC2 / 32) / 8; // 4608 (2 N-tiles/wave)
  const int tobfBlocks = ((int)NX + 255) / 256;

  for (int l = 0; l < kL; ++l) {
    // c = relu(bn1(conv1d(c)))
    k_conv1d<<<tobfBlocks, 256, 0, stream>>>(c_, c1dw + (size_t)l * kDim * kDim * 3, ct_);
    k_bn1<<<kDim, 128, 0, stream>>>(ct_, bn1g + l * kDim, bn1b + l * kDim, c_);
    // x = LN(x + c^T)
    k_ln<<<kB * kN, 128, 0, stream>>>(x_, c_, lng + l * kDim, lnb + l * kDim);
    // c2 += seq2pair(c); emit padded bf16 copy for conv
    k_pair<<<pairBlocks, 256, 0, stream>>>(c2, c2bf, c_, 1);
    // c2 = relu(bn2(conv2d(c2)))   [WMMA implicit GEMM, 2 tiles/wave]
    k_conv2_wmma<<<convBlocks, 256, 0, stream>>>(c2bf, wpkc + (size_t)l * wpkcPer, c2t);
    k_bn2<<<kC2, 256, 0, stream>>>(c2t, bn2g + l * kC2, bn2b + l * kC2, c2);
    // gate from dsym
    k_gate<<<(int)(((size_t)kB * kH * kN * kN + 255) / 256), 256, 0, stream>>>(
        c2, wdw + (size_t)l * kH * kC2, wdb + l * kH, gate);
    // q,k,v = relu(x @ W^T + b)  [WMMA]
    k_tobf<<<tobfBlocks, 256, 0, stream>>>(x_, nullptr, xbf, (int)NX, 0);
    k_gemm_wmma<<<24, 256, 0, stream>>>(xbf, wpkl + (l * 4 + 0) * 8192, wqb + l * kDim, xq, 1);
    k_gemm_wmma<<<24, 256, 0, stream>>>(xbf, wpkl + (l * 4 + 1) * 8192, wkb + l * kDim, xk, 1);
    k_gemm_wmma<<<24, 256, 0, stream>>>(xbf, wpkl + (l * 4 + 2) * 8192, wvb + l * kDim, xv, 1);
    // attention with mask + gate
    k_attn<<<kB * kH * kN, kN, 0, stream>>>(xq, xk, xv, gate, maskp, updb);
    // x = relu((x + upd) @ wl^T + b)  [WMMA]
    k_tobf<<<tobfBlocks, 256, 0, stream>>>(x_, updb, xbf, (int)NX, 1);
    k_gemm_wmma<<<24, 256, 0, stream>>>(xbf, wpkl + (l * 4 + 3) * 8192, wlb + l * kDim, x_, 1);
  }
  // out = relu(x @ fc^T + b)  [WMMA]
  k_tobf<<<tobfBlocks, 256, 0, stream>>>(x_, nullptr, xbf, (int)NX, 0);
  k_gemm_wmma<<<24, 256, 0, stream>>>(xbf, wpkl + 12 * 8192, fcb, (float*)d_out, 1);
}
